// FoldAttention_72043781423279
// MI455X (gfx1250) — matrix operand order repaired and checked by reading, not compiled, not run on hardware
//
#include <hip/hip_runtime.h>
#include <hip/hip_bf16.h>

typedef __attribute__((ext_vector_type(16))) __bf16     v16bf;
typedef __attribute__((ext_vector_type(8)))  float      v8f;
typedef __attribute__((ext_vector_type(4)))  unsigned   u32x4;

union Frag16 { v16bf v; u32x4 q[2]; };

#define IMG    256
#define CIN    64
#define NPATCH 1024
#define LSTR   40   // LDS row stride in bf16 (80B -> every b128 slot 16B-aligned)

// ---------------------------------------------------------------- convert
__global__ __launch_bounds__(256) void cvt_f32_bf16(const float* __restrict__ s,
                                                    __bf16* __restrict__ d, size_t n) {
    size_t i = (size_t)blockIdx.x * blockDim.x + threadIdx.x;
    size_t stride = (size_t)gridDim.x * blockDim.x;
    for (; i < n; i += stride) d[i] = (__bf16)s[i];
}

// ---------------------------------------------------------------- unfold + LN (+affine) -> bf16
__global__ __launch_bounds__(256) void ln_unfold_kernel(
    const float* __restrict__ pan, const float* __restrict__ ms,
    const float* __restrict__ g_q,  const float* __restrict__ b_q,
    const float* __restrict__ g_kp, const float* __restrict__ b_kp,
    const float* __restrict__ g_vp, const float* __restrict__ b_vp,
    const float* __restrict__ g_km, const float* __restrict__ b_km,
    const float* __restrict__ g_vm, const float* __restrict__ b_vm,
    __bf16* __restrict__ Xq,  __bf16* __restrict__ Xkp, __bf16* __restrict__ Xvp,
    __bf16* __restrict__ Xkm, __bf16* __restrict__ Xvm)
{
    const int n = blockIdx.x, which = blockIdx.y, b = blockIdx.z;
    const int L  = (which == 0) ? 8192 : 4096;
    const int hp = n >> 5, wp = n & 31;
    const float* base_p = pan + (size_t)b * CIN * IMG * IMG;
    const float* base_m = ms  + (size_t)b * CIN * IMG * IMG;

    auto fetch = [&](int k) -> float {
        int c = k >> 6, ki = (k >> 3) & 7, kj = k & 7;
        const float* src;
        if (which == 0) { src = (c < 64) ? base_p : base_m; c &= 63; }
        else if (which == 1) src = base_p;
        else                 src = base_m;
        return src[(size_t)c * IMG * IMG + (hp * 8 + ki) * IMG + (wp * 8 + kj)];
    };

    float s = 0.f, ss = 0.f;
    for (int k = threadIdx.x; k < L; k += 256) { float x = fetch(k); s += x; ss += x * x; }
    __shared__ float r1[256], r2[256];
    r1[threadIdx.x] = s; r2[threadIdx.x] = ss; __syncthreads();
    for (int off = 128; off > 0; off >>= 1) {
        if ((int)threadIdx.x < off) { r1[threadIdx.x] += r1[threadIdx.x + off];
                                      r2[threadIdx.x] += r2[threadIdx.x + off]; }
        __syncthreads();
    }
    const float mu  = r1[0] / (float)L;
    const float var = r2[0] / (float)L - mu * mu;
    const float rs  = rsqrtf(var + 1e-5f);
    const size_t row = (size_t)b * NPATCH + n;

    for (int k = threadIdx.x; k < L; k += 256) {
        float y = (fetch(k) - mu) * rs;
        if (which == 0) {
            Xq[row * 8192 + k] = (__bf16)(y * g_q[k] + b_q[k]);
        } else if (which == 1) {
            Xkp[row * 4096 + k] = (__bf16)(y * g_kp[k] + b_kp[k]);
            Xvp[row * 4096 + k] = (__bf16)(y * g_vp[k] + b_vp[k]);
        } else {
            Xkm[row * 4096 + k] = (__bf16)(y * g_km[k] + b_km[k]);
            Xvm[row * 4096 + k] = (__bf16)(y * g_vm[k] + b_vm[k]);
        }
    }
}

// ---------------------------------------------------------------- bf16 GEMM  C[b] = A[b] @ W
// 128x128 tile / workgroup; 8 waves in 4(M) x 2(N) arrangement, each wave
// 32(M) x 64(N) = 2 A-frags x 4 B-frags -> 8 WMMAs with operand reuse.
// LDS tiles double-buffered; next tile prefetched to registers before barrier.
__global__ __launch_bounds__(256) void gemm_bf16_kernel(
    const __bf16* __restrict__ A, size_t strideAb, int lda,
    const __bf16* __restrict__ W, int N, int K,
    __bf16* __restrict__ C, size_t strideCb)
{
    __shared__ __bf16 At[2][128][LSTR];
    __shared__ __bf16 Bt[2][128][LSTR];   // transposed: Bt[buf][n][k]

    const int b = blockIdx.z;
    const int mbase = blockIdx.y * 128, nbase = blockIdx.x * 128;
    const int tid = threadIdx.x, wave = tid >> 5, lane = tid & 31;
    const int l16 = lane & 15, h = lane >> 4;
    const int waveM = wave & 3, waveN = wave >> 2;
    const __bf16* Ab = A + (size_t)b * strideAb;

    // staging coordinates
    const int arow = tid >> 1, acol = (tid & 1) * 16;          // A: 128x32
    const int wk   = (tid * 16) >> 7, wcol = (tid * 16) & 127; // W: 32x128

    v8f zero = {0.f,0.f,0.f,0.f,0.f,0.f,0.f,0.f};
    v8f acc[2][4];
#pragma unroll
    for (int i = 0; i < 2; i++)
#pragma unroll
        for (int t = 0; t < 4; t++) acc[i][t] = zero;

    auto loadA = [&](int kk, u32x4& a0, u32x4& a1) {
        const u32x4* src = (const u32x4*)(Ab + (size_t)(mbase + arow) * lda + kk + acol);
        a0 = src[0]; a1 = src[1];
    };
    auto loadW = [&](int kk, Frag16& wv) {
        const __bf16* wp = W + (size_t)(kk + wk) * N + nbase + wcol;
        wv.q[0] = *(const u32x4*)wp;
        wv.q[1] = *(const u32x4*)(wp + 8);
    };
    auto storeA = [&](int buf, u32x4 a0, u32x4 a1) {
        *(u32x4*)&At[buf][arow][acol]     = a0;
        *(u32x4*)&At[buf][arow][acol + 8] = a1;
    };
    auto storeW = [&](int buf, const Frag16& wv) {
#pragma unroll
        for (int j = 0; j < 16; j++) Bt[buf][wcol + j][wk] = wv.v[j];
    };

    // prologue: stage first tile into buffer 0
    u32x4 a0, a1; Frag16 wv;
    loadA(0, a0, a1); loadW(0, wv);
    storeA(0, a0, a1); storeW(0, wv);

    const int nsteps = K >> 5;
    for (int s = 0; s < nsteps; s++) {
        const int buf = s & 1;
        const bool hasNext = (s + 1 < nsteps);
        if (hasNext) { loadA((s + 1) << 5, a0, a1); loadW((s + 1) << 5, wv); }

        __syncthreads();   // tile in buf is ready for everyone

        // gather all fragments first (single dscnt drain), then 8 WMMAs
        Frag16 af[2], bf_[4];
#pragma unroll
        for (int i = 0; i < 2; i++) {
            const int r = 32 * waveM + 16 * i + l16;
            af[i].q[0] = *(const u32x4*)&At[buf][r][h * 8];
            af[i].q[1] = *(const u32x4*)&At[buf][r][16 + h * 8];
        }
#pragma unroll
        for (int t = 0; t < 4; t++) {
            const int c = 64 * waveN + 16 * t + l16;
            bf_[t].q[0] = *(const u32x4*)&Bt[buf][c][h * 8];
            bf_[t].q[1] = *(const u32x4*)&Bt[buf][c][16 + h * 8];
        }
#pragma unroll
        for (int i = 0; i < 2; i++)
#pragma unroll
            for (int t = 0; t < 4; t++)
                acc[i][t] = __builtin_amdgcn_wmma_f32_16x16x32_bf16(
                    false, af[i].v, false, bf_[t].v, (short)0, acc[i][t], false, false);

        if (hasNext) { storeA(buf ^ 1, a0, a1); storeW(buf ^ 1, wv); }
    }

#pragma unroll
    for (int i = 0; i < 2; i++)
#pragma unroll
        for (int t = 0; t < 4; t++)
#pragma unroll
            for (int r = 0; r < 8; r++) {
                int row = mbase + 32 * waveM + 16 * i + (h ? 8 + r : r);
                int col = nbase + 64 * waveN + 16 * t + l16;
                C[(size_t)b * strideCb + (size_t)row * N + col] = (__bf16)acc[i][t][r];
            }
}

// ---------------------------------------------------------------- fused flash attention + fold + residual
__global__ __launch_bounds__(256) void attn_kernel(
    const __bf16* __restrict__ Q,                                  // [2,1024,1024]
    const __bf16* __restrict__ Kp, const __bf16* __restrict__ Km,  // [2,1024,1024]
    const __bf16* __restrict__ Vp, const __bf16* __restrict__ Vm,  // [2,1024,4096]
    const float* __restrict__ pan, const float* __restrict__ ms,
    float* __restrict__ out)                                       // [2,128,256,256]
{
    const int mtile = blockIdx.x;            // 0..7  (128 query rows)
    const int g     = blockIdx.y;            // head 0..31
    const int z     = blockIdx.z;            // 0..3
    const int b = z & 1, src = z >> 1;
    const __bf16* Kh    = src ? Km : Kp;
    const __bf16* Vh    = src ? Vm : Vp;
    const float*  resid = src ? ms : pan;

    __shared__ __bf16 Vt[2][128][LSTR];      // V block transposed, double-buffered
    __shared__ __bf16 Pl[8][16][LSTR];       // per-wave softmax probs (16x32)

    const int tid = threadIdx.x, wave = tid >> 5, lane = tid & 31;
    const int l16 = lane & 15, h = lane >> 4;
    const int mrowbase = mtile * 128 + wave * 16;
    const float SCALE = 0.08838834764831845f;   // (4096/32)^-0.5

    // Q fragment, loaded once (rows mrowbase..+15, cols g*32..+31)
    const __bf16* Qp = Q + ((size_t)b * NPATCH + mrowbase + l16) * 1024 + g * 32;
    Frag16 qf;
    qf.q[0] = *(const u32x4*)(Qp + h * 8);
    qf.q[1] = *(const u32x4*)(Qp + 16 + h * 8);

    float mrow[8], lrow[8];
#pragma unroll
    for (int r = 0; r < 8; r++) { mrow[r] = -1e30f; lrow[r] = 0.f; }
    v8f zero = {0.f,0.f,0.f,0.f,0.f,0.f,0.f,0.f};
    v8f acc[8];
#pragma unroll
    for (int t = 0; t < 8; t++) acc[t] = zero;

    const __bf16* Kbase = Kh + (size_t)b * NPATCH * 1024 + g * 32;
    const __bf16* Vbase = Vh + (size_t)b * NPATCH * 4096 + g * 128;

    // V staging coords: 32 rows x 128 cols per block, stored transposed
    const int vrow = tid >> 3, vcolh = (tid & 7) * 16;
    auto loadV = [&](int kb, Frag16& vv) {
        const __bf16* vp = Vbase + (size_t)(kb + vrow) * 4096 + vcolh;
        vv.q[0] = *(const u32x4*)vp;
        vv.q[1] = *(const u32x4*)(vp + 8);
    };
    auto storeV = [&](int buf, const Frag16& vv) {
#pragma unroll
        for (int j = 0; j < 16; j++) Vt[buf][vcolh + j][vrow] = vv.v[j];
    };

    // prologue: stage V block 0
    Frag16 vv;
    loadV(0, vv); storeV(0, vv);

    for (int s = 0; s < 32; s++) {
        const int kb  = s << 5;
        const int buf = s & 1;
        const bool hasNext = (s + 1 < 32);
        if (hasNext) loadV(kb + 32, vv);

        // S = Q @ K^T for this 32-key block (K fragments straight from global)
        Frag16 kf0, kf1;
        const __bf16* kp0 = Kbase + (size_t)(kb + l16) * 1024 + h * 8;
        const __bf16* kp1 = Kbase + (size_t)(kb + 16 + l16) * 1024 + h * 8;
        kf0.q[0] = *(const u32x4*)kp0; kf0.q[1] = *(const u32x4*)(kp0 + 16);
        kf1.q[0] = *(const u32x4*)kp1; kf1.q[1] = *(const u32x4*)(kp1 + 16);
        v8f s0 = __builtin_amdgcn_wmma_f32_16x16x32_bf16(false, qf.v, false, kf0.v, (short)0, zero, false, false);
        v8f s1 = __builtin_amdgcn_wmma_f32_16x16x32_bf16(false, qf.v, false, kf1.v, (short)0, zero, false, false);

#pragma unroll
        for (int r = 0; r < 8; r++) {
            float a0 = s0[r] * SCALE, a1 = s1[r] * SCALE;
            float mv = fmaxf(a0, a1);
            mv = fmaxf(mv, __shfl_xor(mv, 1, 32));
            mv = fmaxf(mv, __shfl_xor(mv, 2, 32));
            mv = fmaxf(mv, __shfl_xor(mv, 4, 32));
            mv = fmaxf(mv, __shfl_xor(mv, 8, 32));
            float newm  = fmaxf(mrow[r], mv);
            float alpha = __expf(mrow[r] - newm);
            float p0 = __expf(a0 - newm), p1 = __expf(a1 - newm);
            float rsum = p0 + p1;
            rsum += __shfl_xor(rsum, 1, 32);
            rsum += __shfl_xor(rsum, 2, 32);
            rsum += __shfl_xor(rsum, 4, 32);
            rsum += __shfl_xor(rsum, 8, 32);
            lrow[r] = lrow[r] * alpha + rsum;
            mrow[r] = newm;
#pragma unroll
            for (int t = 0; t < 8; t++) acc[t][r] *= alpha;
            int prow = h ? (8 + r) : r;
            Pl[wave][prow][l16]      = (__bf16)p0;
            Pl[wave][prow][16 + l16] = (__bf16)p1;
        }

        __syncthreads();   // Vt[buf] (stored last iteration) visible; P store drained

        // gather P-frag + all 8 V-frags, then 8 WMMAs back-to-back
        Frag16 pf, vf[8];
        pf.q[0] = *(const u32x4*)&Pl[wave][l16][h * 8];
        pf.q[1] = *(const u32x4*)&Pl[wave][l16][16 + h * 8];
#pragma unroll
        for (int t = 0; t < 8; t++) {
            vf[t].q[0] = *(const u32x4*)&Vt[buf][16 * t + l16][h * 8];
            vf[t].q[1] = *(const u32x4*)&Vt[buf][16 * t + l16][16 + h * 8];
        }
#pragma unroll
        for (int t = 0; t < 8; t++)
            acc[t] = __builtin_amdgcn_wmma_f32_16x16x32_bf16(
                false, pf.v, false, vf[t].v, (short)0, acc[t], false, false);

        if (hasNext) storeV(buf ^ 1, vv);
    }

    // epilogue: normalize, fold-scatter, residual add
    float inv[8];
#pragma unroll
    for (int r = 0; r < 8; r++) inv[r] = 1.f / lrow[r];
#pragma unroll
    for (int t = 0; t < 8; t++)
#pragma unroll
        for (int r = 0; r < 8; r++) {
            int n  = mrowbase + (h ? 8 + r : r);
            int cc = 16 * t + l16;
            int f  = g * 128 + cc;
            int c  = f >> 6, ki = (f >> 3) & 7, kj = f & 7;
            int hp = n >> 5, wp = n & 31;
            size_t sp = (size_t)(hp * 8 + ki) * IMG + (wp * 8 + kj);
            float val = acc[t][r] * inv[r];
            float res = resid[((size_t)b * 64 + c) * (IMG * IMG) + sp];
            out[((size_t)b * 128 + src * 64 + c) * (IMG * IMG) + sp] = val + res;
        }
}

// ---------------------------------------------------------------- launch
extern "C" void kernel_launch(void* const* d_in, const int* in_sizes, int n_in,
                              void* d_out, int out_size, void* d_ws, size_t ws_size,
                              hipStream_t stream) {
    const float* pan  = (const float*)d_in[0];
    const float* ms   = (const float*)d_in[1];
    const float* g_q  = (const float*)d_in[2];
    const float* b_q  = (const float*)d_in[3];
    const float* g_kp = (const float*)d_in[4];
    const float* b_kp = (const float*)d_in[5];
    const float* g_vp = (const float*)d_in[6];
    const float* b_vp = (const float*)d_in[7];
    const float* g_km = (const float*)d_in[8];
    const float* b_km = (const float*)d_in[9];
    const float* g_vm = (const float*)d_in[10];
    const float* b_vm = (const float*)d_in[11];
    const float* w_q  = (const float*)d_in[12];
    const float* w_kp = (const float*)d_in[13];
    const float* w_vp = (const float*)d_in[14];
    const float* w_km = (const float*)d_in[15];
    const float* w_vm = (const float*)d_in[16];
    float* out = (float*)d_out;

    char* ws = (char*)d_ws;
    size_t off = 0;
    auto bump = [&](size_t elems) -> __bf16* {
        __bf16* p = (__bf16*)(ws + off);
        off += ((elems * 2 + 255) & ~(size_t)255);
        return p;
    };
    __bf16* Xq  = bump(2ull * 1024 * 8192);
    __bf16* Xkp = bump(2ull * 1024 * 4096);
    __bf16* Xvp = bump(2ull * 1024 * 4096);
    __bf16* Xkm = bump(2ull * 1024 * 4096);
    __bf16* Xvm = bump(2ull * 1024 * 4096);
    __bf16* Wq  = bump(8192ull * 1024);
    __bf16* Wkp = bump(4096ull * 1024);
    __bf16* Wvp = bump(4096ull * 4096);
    __bf16* Wkm = bump(4096ull * 1024);
    __bf16* Wvm = bump(4096ull * 4096);
    __bf16* Qb  = bump(2ull * 1024 * 1024);
    __bf16* Kpb = bump(2ull * 1024 * 1024);
    __bf16* Kmb = bump(2ull * 1024 * 1024);
    __bf16* Vpb = bump(2ull * 1024 * 4096);
    __bf16* Vmb = bump(2ull * 1024 * 4096);

    // 1) weights -> bf16
    cvt_f32_bf16<<<1024, 256, 0, stream>>>(w_q,  Wq,  8192ull * 1024);
    cvt_f32_bf16<<<1024, 256, 0, stream>>>(w_kp, Wkp, 4096ull * 1024);
    cvt_f32_bf16<<<1024, 256, 0, stream>>>(w_vp, Wvp, 4096ull * 4096);
    cvt_f32_bf16<<<1024, 256, 0, stream>>>(w_km, Wkm, 4096ull * 1024);
    cvt_f32_bf16<<<1024, 256, 0, stream>>>(w_vm, Wvm, 4096ull * 4096);

    // 2) unfold + LN (+affine) -> bf16 activations
    ln_unfold_kernel<<<dim3(1024, 3, 2), 256, 0, stream>>>(
        pan, ms, g_q, b_q, g_kp, b_kp, g_vp, b_vp, g_km, b_km, g_vm, b_vm,
        Xq, Xkp, Xvp, Xkm, Xvm);

    // 3) projections (batched WMMA GEMMs)
    gemm_bf16_kernel<<<dim3(8, 8, 2),  256, 0, stream>>>(Xq,  1024ull*8192, 8192, Wq,  1024, 8192, Qb,  1024ull*1024);
    gemm_bf16_kernel<<<dim3(8, 8, 2),  256, 0, stream>>>(Xkp, 1024ull*4096, 4096, Wkp, 1024, 4096, Kpb, 1024ull*1024);
    gemm_bf16_kernel<<<dim3(8, 8, 2),  256, 0, stream>>>(Xkm, 1024ull*4096, 4096, Wkm, 1024, 4096, Kmb, 1024ull*1024);
    gemm_bf16_kernel<<<dim3(32, 8, 2), 256, 0, stream>>>(Xvp, 1024ull*4096, 4096, Wvp, 4096, 4096, Vpb, 1024ull*4096);
    gemm_bf16_kernel<<<dim3(32, 8, 2), 256, 0, stream>>>(Xvm, 1024ull*4096, 4096, Wvm, 4096, 4096, Vmb, 1024ull*4096);

    // 4) fused attention + softmax + fold + residual
    attn_kernel<<<dim3(8, 32, 4), 256, 0, stream>>>(Qb, Kpb, Kmb, Vpb, Vmb, pan, ms, out);
}
